// KoopRNN_Block_19456201851453
// MI455X (gfx1250) — matrix operand order repaired and checked by reading, not compiled, not run on hardware
//
#include <hip/hip_runtime.h>
#include <hip/hip_bf16.h>

typedef __bf16 bf16;
typedef __attribute__((ext_vector_type(16))) __bf16 v16bf;
typedef __attribute__((ext_vector_type(8)))  __bf16 v8bf;
typedef __attribute__((ext_vector_type(8)))  float  v8f;
typedef __attribute__((ext_vector_type(4)))  float  v4f;

#define NBLK   8
#define BATCH  8192
#define SEGN   4      // FREQ
#define STEPN  4
#define DYN    128
#define HID    256
#define IND    192    // SEG_LEN * ENC_IN
#define ROWLEN 768    // INPUT_LEN * ENC_IN

// ---------------- WMMA helpers (CDNA5, wave32) ----------------

__device__ __forceinline__ v8f wmma_bf16(v16bf a, v16bf b, v8f c) {
  // 8 args: (neg_a, A, neg_b, B, c_mod, C, reuse_a, reuse_b)
  return __builtin_amdgcn_wmma_f32_16x16x32_bf16(false, a, false, b, (short)0, c,
                                                 false, false);
}

// A fragment 16x32 bf16 from row-major [16][lda] LDS buffer, k-offset k0.
// Lanes 0-15 hold M=lane, K=k0..k0+15;
//             lanes 16-31 hold M=lane-16, K=k0+16..k0+31 (matches load_B).
__device__ __forceinline__ v16bf load_A(const bf16* A, int lda, int lane, int k0) {
  int m   = lane & 15;
  int khi = (lane >> 4) << 4;                  // 0 or 16
  const bf16* base = A + m * lda + k0 + khi;
  v8bf lo = *(const v8bf*)(base);              // K = k0+khi .. +7   (16B aligned)
  v8bf hi = *(const v8bf*)(base + 8);          // K = k0+khi+8 .. +15
  return __builtin_shufflevector(lo, hi, 0,1,2,3,4,5,6,7,8,9,10,11,12,13,14,15);
}

// B fragment 32x16 (KxN) from row-major weight W[N][K] bf16 in global (L2-hot).
// Lane l<16: column n=l, K=k0..k0+15; lane l>=16: column n=l-16, K=k0+16..k0+31.
// One contiguous 32-byte load per lane.
__device__ __forceinline__ v16bf load_B(const bf16* W, int K, int ntile, int lane, int k0) {
  int n   = lane & 15;
  int khi = (lane >> 4) << 4;                  // 0 or 16
  return *(const v16bf*)(W + (size_t)(ntile * 16 + n) * K + k0 + khi);
}

// ---------------- weight f32 -> bf16 pre-pass ----------------

__global__ void f32_to_bf16(const float* __restrict__ src, bf16* __restrict__ dst, int n) {
  int i = blockIdx.x * blockDim.x + threadIdx.x;
  int stride = gridDim.x * blockDim.x;
  for (; i < n; i += stride) dst[i] = (bf16)src[i];
}

// ---------------- fused KoopRNN block kernel ----------------
// LDS regions (byte offsets), overlapped by lifetime:
//   [0      , 32768) : segb(24KB, st0-1) | preds f32(32KB, st4-5) | d1(32KB, st6-7)
//   [32768  , 81920) : h1(32KB, st1-2)   | nrm(16KB, st5-6)       | obuf f32(48KB, st7-copy)
//   [81920  , 98304) : xe(16KB, st2-3)
//   [98304  ,106496) : h double-buffer (8KB, st3-4)
//   [106496 ,107008) : mu/rstd (64 rows each)
#define SMEM_BYTES 107008

__global__ __launch_bounds__(256)
void koop_fused(const float* __restrict__ x,
                const bf16*  __restrict__ w_e1, const float* __restrict__ b_e1,
                const bf16*  __restrict__ w_e2, const float* __restrict__ b_e2,
                const bf16*  __restrict__ w_xh, const bf16*  __restrict__ w_hh,
                const float* __restrict__ lng,  const float* __restrict__ lnb,
                const bf16*  __restrict__ w_d1, const float* __restrict__ b_d1,
                const bf16*  __restrict__ w_d2, const float* __restrict__ b_d2,
                float* __restrict__ out)
{
  extern __shared__ char smem[];
  const int tid   = threadIdx.x;
  const int lane  = tid & 31;
  const int wave  = tid >> 5;
  const int nloc  = lane & 15;
  const int mbase = (lane >> 4) << 3;

  const int nb   = blockIdx.y;
  const int row0 = blockIdx.x * 16;

  const bf16*  W1  = w_e1 + (size_t)nb * HID * IND;
  const float* B1  = b_e1 + nb * HID;
  const bf16*  W2  = w_e2 + (size_t)nb * DYN * HID;
  const float* B2  = b_e2 + nb * DYN;
  const bf16*  WX  = w_xh + (size_t)nb * DYN * DYN;
  const bf16*  WH  = w_hh + (size_t)nb * DYN * DYN;
  const float* G   = lng  + nb * DYN;
  const float* Bt  = lnb  + nb * DYN;
  const bf16*  D1  = w_d1 + (size_t)nb * HID * DYN;
  const float* Bd1 = b_d1 + nb * HID;
  const bf16*  D2  = w_d2 + (size_t)nb * IND * HID;
  const float* Bd2 = b_d2 + nb * IND;

  bf16*  segb  = (bf16*)(smem + 0);       // [16][768]
  float* preds = (float*)(smem + 0);      // [4*16][128]
  bf16*  d1b   = (bf16*)(smem + 0);       // [4*16][256]
  bf16*  h1b   = (bf16*)(smem + 32768);   // [4*16][256]
  bf16*  nrm   = (bf16*)(smem + 32768);   // [4*16][128]
  float* obuf  = (float*)(smem + 32768);  // [16][768]
  bf16*  xeb   = (bf16*)(smem + 81920);   // [4*16][128]
  bf16*  hb    = (bf16*)(smem + 98304);   // [2][16][128]
  float* muv   = (float*)(smem + 106496); // [64]
  float* rsv   = (float*)(smem + 106752); // [64]

  // ---- stage 0: load 16 input rows (768 f32 each), convert to bf16 in LDS ----
  {
    const float* xin = x + ((size_t)nb * BATCH + row0) * ROWLEN;
    for (int i = tid; i < 16 * ROWLEN / 4; i += 256) {
      v4f v = *(const v4f*)(xin + i * 4);
      bf16* d = segb + i * 4;
      d[0] = (bf16)v.x; d[1] = (bf16)v.y; d[2] = (bf16)v.z; d[3] = (bf16)v.w;
    }
  }
  __syncthreads();

  // ---- stage 1: encoder L1: relu(seg @ W1^T + b1) -> h1 (4 segs, 16x256) ----
  // B fragments register-cached per ntile, reused across the 4 segments.
  for (int nt = wave; nt < HID / 16; nt += 8) {
    v16bf Bf[6];
    #pragma unroll
    for (int kk = 0; kk < 6; ++kk) Bf[kk] = load_B(W1, IND, nt, lane, kk * 32);
    float bias = B1[nt * 16 + nloc];
    #pragma unroll
    for (int seg = 0; seg < SEGN; ++seg) {
      v8f acc;
      #pragma unroll
      for (int r = 0; r < 8; ++r) acc[r] = bias;
      const bf16* A = segb + seg * IND;             // lda = ROWLEN
      #pragma unroll
      for (int kk = 0; kk < 6; ++kk)
        acc = wmma_bf16(load_A(A, ROWLEN, lane, kk * 32), Bf[kk], acc);
      #pragma unroll
      for (int r = 0; r < 8; ++r) {
        float v = acc[r]; v = v > 0.f ? v : 0.f;
        h1b[(seg * 16 + mbase + r) * HID + nt * 16 + nloc] = (bf16)v;
      }
    }
  }
  __syncthreads();

  // ---- stage 2: encoder L2: h1 @ W2^T + b2 -> x_enc (4 segs, 16x128) ----
  {
    int nt = wave;                                  // 8 ntiles == 8 waves
    v16bf Bf[8];
    #pragma unroll
    for (int kk = 0; kk < 8; ++kk) Bf[kk] = load_B(W2, HID, nt, lane, kk * 32);
    float bias = B2[nt * 16 + nloc];
    #pragma unroll
    for (int seg = 0; seg < SEGN; ++seg) {
      v8f acc;
      #pragma unroll
      for (int r = 0; r < 8; ++r) acc[r] = bias;
      const bf16* A = h1b + seg * 16 * HID;
      #pragma unroll
      for (int kk = 0; kk < 8; ++kk)
        acc = wmma_bf16(load_A(A, HID, lane, kk * 32), Bf[kk], acc);
      #pragma unroll
      for (int r = 0; r < 8; ++r)
        xeb[(seg * 16 + mbase + r) * DYN + nt * 16 + nloc] = (bf16)acc[r];
    }
  }
  __syncthreads();

  // ---- stages 3+4: RNN recurrence + free-running prediction ----
  // Wxh fragments used 4x, Whh fragments used 7x: cache both in registers.
  {
    int nt = wave;
    v16bf BX[4], BH[4];
    #pragma unroll
    for (int kk = 0; kk < 4; ++kk) BX[kk] = load_B(WX, DYN, nt, lane, kk * 32);
    #pragma unroll
    for (int kk = 0; kk < 4; ++kk) BH[kk] = load_B(WH, DYN, nt, lane, kk * 32);

    // recurrence: h_t = x_t @ Wxh^T + h_{t-1} @ Whh^T
    for (int seg = 0; seg < SEGN; ++seg) {
      v8f acc;
      #pragma unroll
      for (int r = 0; r < 8; ++r) acc[r] = 0.f;
      const bf16* A = xeb + seg * 16 * DYN;
      #pragma unroll
      for (int kk = 0; kk < 4; ++kk)
        acc = wmma_bf16(load_A(A, DYN, lane, kk * 32), BX[kk], acc);
      if (seg > 0) {
        const bf16* Hp = hb + ((seg + 1) & 1) * 16 * DYN;
        #pragma unroll
        for (int kk = 0; kk < 4; ++kk)
          acc = wmma_bf16(load_A(Hp, DYN, lane, kk * 32), BH[kk], acc);
      }
      bf16* Hw = hb + (seg & 1) * 16 * DYN;
      #pragma unroll
      for (int r = 0; r < 8; ++r)
        Hw[(mbase + r) * DYN + nt * 16 + nloc] = (bf16)acc[r];
      __syncthreads();
    }

    // free-running: h <- h @ Whh^T, collect preds (f32)
    for (int s = 0; s < STEPN; ++s) {
      const bf16* Hp = hb + ((s + 1) & 1) * 16 * DYN;  // final RNN h is in buffer 1
      v8f acc;
      #pragma unroll
      for (int r = 0; r < 8; ++r) acc[r] = 0.f;
      #pragma unroll
      for (int kk = 0; kk < 4; ++kk)
        acc = wmma_bf16(load_A(Hp, DYN, lane, kk * 32), BH[kk], acc);
      bf16* Hw = hb + (s & 1) * 16 * DYN;
      #pragma unroll
      for (int r = 0; r < 8; ++r) {
        Hw[(mbase + r) * DYN + nt * 16 + nloc] = (bf16)acc[r];
        preds[(s * 16 + mbase + r) * DYN + nt * 16 + nloc] = acc[r];
      }
      __syncthreads();
    }
  }

  // ---- stage 5: LayerNorm over DYN, affine, -> bf16 nrm ----
  if (tid < 64) {
    const float* p = preds + tid * DYN;
    float s = 0.f, sq = 0.f;
    for (int j = 0; j < DYN; ++j) { float v = p[j]; s += v; sq += v * v; }
    float mu  = s * (1.f / 128.f);
    float var = sq * (1.f / 128.f) - mu * mu;
    muv[tid] = mu;
    rsv[tid] = rsqrtf(var + 1e-5f);
  }
  __syncthreads();
  for (int i = tid; i < 64 * DYN; i += 256) {
    int rid = i >> 7, j = i & 127;
    float v = (preds[i] - muv[rid]) * rsv[rid] * G[j] + Bt[j];
    nrm[i] = (bf16)v;
  }
  __syncthreads();

  // ---- stage 6: decoder L1: relu(nrm @ D1^T + b) -> d1 (4 steps, 16x256) ----
  for (int nt = wave; nt < HID / 16; nt += 8) {
    v16bf Bf[4];
    #pragma unroll
    for (int kk = 0; kk < 4; ++kk) Bf[kk] = load_B(D1, DYN, nt, lane, kk * 32);
    float bias = Bd1[nt * 16 + nloc];
    #pragma unroll
    for (int s = 0; s < STEPN; ++s) {
      v8f acc;
      #pragma unroll
      for (int r = 0; r < 8; ++r) acc[r] = bias;
      const bf16* A = nrm + s * 16 * DYN;
      #pragma unroll
      for (int kk = 0; kk < 4; ++kk)
        acc = wmma_bf16(load_A(A, DYN, lane, kk * 32), Bf[kk], acc);
      #pragma unroll
      for (int r = 0; r < 8; ++r) {
        float v = acc[r]; v = v > 0.f ? v : 0.f;
        d1b[(s * 16 + mbase + r) * HID + nt * 16 + nloc] = (bf16)v;
      }
    }
  }
  __syncthreads();

  // ---- stage 7: decoder L2: d1 @ D2^T + b -> obuf f32 [16][768] ----
  for (int nt = wave; nt < IND / 16; nt += 8) {   // 12 ntiles; waves 0-3 take 2
    v16bf Bf[8];
    #pragma unroll
    for (int kk = 0; kk < 8; ++kk) Bf[kk] = load_B(D2, HID, nt, lane, kk * 32);
    float bias = Bd2[nt * 16 + nloc];
    #pragma unroll
    for (int s = 0; s < STEPN; ++s) {
      v8f acc;
      #pragma unroll
      for (int r = 0; r < 8; ++r) acc[r] = bias;
      const bf16* A = d1b + s * 16 * HID;
      #pragma unroll
      for (int kk = 0; kk < 8; ++kk)
        acc = wmma_bf16(load_A(A, HID, lane, kk * 32), Bf[kk], acc);
      #pragma unroll
      for (int r = 0; r < 8; ++r)
        obuf[(mbase + r) * ROWLEN + s * IND + nt * 16 + nloc] = acc[r];
    }
  }
  __syncthreads();

  // ---- final: coalesced f32 store; (s,p,c) layout is already contiguous per row ----
  {
    float* oo = out + ((size_t)nb * BATCH + row0) * ROWLEN;
    for (int i = tid; i < 16 * ROWLEN / 4; i += 256)
      *(v4f*)(oo + i * 4) = *(const v4f*)(obuf + i * 4);
  }
}

// ---------------- launch ----------------

extern "C" void kernel_launch(void* const* d_in, const int* in_sizes, int n_in,
                              void* d_out, int out_size, void* d_ws, size_t ws_size,
                              hipStream_t stream) {
  const float* x    = (const float*)d_in[0];
  const float* ew1  = (const float*)d_in[1];
  const float* eb1  = (const float*)d_in[2];
  const float* ew2  = (const float*)d_in[3];
  const float* eb2  = (const float*)d_in[4];
  const float* wxh  = (const float*)d_in[5];
  const float* whh  = (const float*)d_in[6];
  const float* lng  = (const float*)d_in[7];
  const float* lnb  = (const float*)d_in[8];
  const float* dw1  = (const float*)d_in[9];
  const float* db1  = (const float*)d_in[10];
  const float* dw2  = (const float*)d_in[11];
  const float* db2  = (const float*)d_in[12];
  float* out = (float*)d_out;

  // bf16 weight staging in workspace (~3 MB)
  bf16* ws = (bf16*)d_ws;
  const int n_w1 = NBLK * HID * IND;   // 393216
  const int n_w2 = NBLK * DYN * HID;   // 262144
  const int n_wx = NBLK * DYN * DYN;   // 131072
  const int n_d1 = NBLK * HID * DYN;   // 262144
  const int n_d2 = NBLK * IND * HID;   // 393216
  bf16* bw1 = ws;
  bf16* bw2 = bw1 + n_w1;
  bf16* bwx = bw2 + n_w2;
  bf16* bwh = bwx + n_wx;
  bf16* bd1 = bwh + n_wx;
  bf16* bd2 = bd1 + n_d1;

  f32_to_bf16<<<256, 256, 0, stream>>>(ew1, bw1, n_w1);
  f32_to_bf16<<<256, 256, 0, stream>>>(ew2, bw2, n_w2);
  f32_to_bf16<<<256, 256, 0, stream>>>(wxh, bwx, n_wx);
  f32_to_bf16<<<256, 256, 0, stream>>>(whh, bwh, n_wx);
  f32_to_bf16<<<256, 256, 0, stream>>>(dw1, bd1, n_d1);
  f32_to_bf16<<<256, 256, 0, stream>>>(dw2, bd2, n_d2);

  dim3 grid(BATCH / 16, NBLK);
  koop_fused<<<grid, 256, SMEM_BYTES, stream>>>(
      x, bw1, eb1, bw2, eb2, bwx, bwh, lng, lnb, bd1, db1, bd2, db2, out);
}
